// DeepIO_2611340116384
// MI455X (gfx1250) — matrix operand order repaired and checked by reading, not compiled, not run on hardware
//
#include <hip/hip_runtime.h>

// ---------------------------------------------------------------------------
// CDNA5 (gfx1250) bf16 WMMA BiLSTM implementation.
//   B=64, T=1024, H=256, gates G=4H=1024, layer-1 input I1=2H=512.
// ---------------------------------------------------------------------------

#define B_  64
#define T_  1024
#define H_  256
#define G_  1024
#define I1_ 512

typedef __attribute__((ext_vector_type(16))) __bf16 v16bf;
typedef __attribute__((ext_vector_type(8)))  __bf16 v8bf;
typedef __attribute__((ext_vector_type(8)))  float  v8f;
typedef __attribute__((ext_vector_type(4)))  float  v4f;

__device__ __forceinline__ v8f wmma_bf16(v16bf a, v16bf b, v8f c) {
    // D = A(16x32 bf16) * B(32x16 bf16) + C(16x16 f32)
    return __builtin_amdgcn_wmma_f32_16x16x32_bf16(
        /*neg_a=*/false, a, /*neg_b=*/false, b,
        /*c_mod=*/(short)0, c, /*reuse_a=*/false, /*reuse_b=*/false);
}

__device__ __forceinline__ v16bf cat8(v8bf lo, v8bf hi) {
    return __builtin_shufflevector(lo, hi, 0,1,2,3,4,5,6,7,8,9,10,11,12,13,14,15);
}

// A fragment (16x32 bf16), source row-major with leading dim `ld` (elements).
// Per ISA 16-bit A layout: lane<16 holds M=lane, K={k0..k0+7, k0+16..k0+23};
// lane>=16 holds M=lane-16, K={k0+8..k0+15, k0+24..k0+31}.
__device__ __forceinline__ v16bf load_a(const __bf16* src, int ld, int k0, int lane) {
    int m  = lane & 15;
    int kh = (lane >> 4) << 3;           // 0 or 8
    const __bf16* p = src + m * ld + k0 + kh;
    v8bf lo = *(const v8bf*)(p);
    v8bf hi = *(const v8bf*)(p + 16);
    return cat8(lo, hi);
}

// Branchless sigmoid/tanh (correct saturation at +/-inf, no EXEC divergence).
__device__ __forceinline__ float sigf(float x) {
    return __builtin_amdgcn_rcpf(1.0f + __expf(-x));
}
__device__ __forceinline__ float tanh_fast(float x) {
    return 2.0f * __builtin_amdgcn_rcpf(1.0f + __expf(-2.0f * x)) - 1.0f;
}

// ---------------------------------------------------------------------------
// Weight packing: W [dirs][G][K] f32 (gate-major, out[n] = sum_k A[k]*W[n][k])
// -> bf16 B-fragments: Wp[((nt*KT + kt)*32 + lane)*16 + e] where
//    N = nt*16 + lane%16, K = kt*32 + (lane/16)*8 + (e%8) + (e/8)*16.
// ---------------------------------------------------------------------------
__global__ void pack_b_kernel(const float* __restrict__ W, __bf16* __restrict__ Wp,
                              int K, int total) {
    int idx = blockIdx.x * blockDim.x + threadIdx.x;
    if (idx >= total) return;
    int perDir = G_ * K;
    int dir = idx / perDir;
    int j   = idx - dir * perDir;
    int e   = j & 15;
    int ln  = (j >> 4) & 31;
    int blk = j >> 9;
    int KT  = K >> 5;
    int kt  = blk % KT;
    int nt  = blk / KT;
    int N   = nt * 16 + (ln & 15);
    int Kc  = kt * 32 + ((ln >> 4) << 3) + (e & 7) + ((e >> 3) << 4);
    Wp[idx] = (__bf16)(W[(size_t)dir * perDir + (size_t)N * K + Kc]);
}

__global__ void bias_kernel(const float* __restrict__ bi0, const float* __restrict__ bh0,
                            const float* __restrict__ bi1, const float* __restrict__ bh1,
                            float* __restrict__ out0, float* __restrict__ out1) {
    int i = blockIdx.x * blockDim.x + threadIdx.x;
    if (i < 2 * G_)            out0[i] = bi0[i] + bh0[i];
    else if (i < 4 * G_)       { int j = i - 2 * G_; out1[j] = bi1[j] + bh1[j]; }
}

// ---------------------------------------------------------------------------
// Layer 0 recurrent LSTM (one block per direction, 1024 threads = 32 waves).
// Per step: gates[64x1024] = h[64x256] @ W_hh^T via WMMA; input part (K=3)
// folded in as scalar FMAs during the gate-nonlinearity update.
// Per-lane gate constants {w0,w1,w2,bias} live in LDS (one ds_load_b128 each)
// to keep the per-wave VGPR count under the 32-wave residency cap (no spills).
// ---------------------------------------------------------------------------
__global__ __launch_bounds__(1024) void lstm_l0_kernel(
    const float*  __restrict__ x,        // [B][T][3]
    const int*    __restrict__ lengths,  // [B]
    const float*  __restrict__ W_ih,     // [2][G][3]
    const __bf16* __restrict__ Wp_hh,    // [2][G*H] packed
    const float*  __restrict__ bias,     // [2][G]
    __bf16*       __restrict__ h1)       // [B][T][512]
{
    const int dir  = blockIdx.x;
    const int tid  = threadIdx.x;
    const int wave = tid >> 5;
    const int lane = tid & 31;
    const int mtile = wave >> 3;          // 0..3
    const int ng    = (wave & 7) * 2;     // h-column tile base: 0,2,..,14

    const float*  Wih = W_ih + dir * (G_ * 3);
    const __bf16* Wp  = Wp_hh + (size_t)dir * (G_ * H_);
    const float*  bb  = bias + dir * G_;

    __shared__ __align__(32) __bf16 hb[B_ * H_];  // 32 KB h-state (bf16)
    __shared__ float xs[2][B_][4];                // double-buffered x staging
    __shared__ v4f   wgb[G_];                     // {w_ih0, w_ih1, w_ih2, bias}

    for (int i = tid; i < B_ * H_; i += 1024) hb[i] = (__bf16)0.0f;
    for (int j = tid; j < G_; j += 1024)
        wgb[j] = (v4f){ Wih[j*3], Wih[j*3+1], Wih[j*3+2], bb[j] };

    float creg[2][8];
    #pragma unroll
    for (int g = 0; g < 2; ++g)
        #pragma unroll
        for (int r = 0; r < 8; ++r) creg[g][r] = 0.0f;

    const int ncol0  = lane & 15;
    const int rowAdd = (lane >> 4) << 3;  // 0 or 8

    // Per-lane lengths for the 8 batch rows this lane updates.
    int Lrow[8];
    #pragma unroll
    for (int r = 0; r < 8; ++r) Lrow[r] = lengths[mtile * 16 + rowAdd + r];

    // Stage x for t = 0.  (Self-contained: nothing stays live across the loop.)
    if (tid < B_) {
        int Lb = lengths[tid];
        int te = (dir == 0) ? 0 : ((0 < Lb) ? (Lb - 1) : 0);
        const float* xp = x + ((size_t)tid * T_ + te) * 3;
        xs[0][tid][0] = xp[0];
        xs[0][tid][1] = xp[1];
        xs[0][tid][2] = xp[2];
    }

    // Per-wave B base: tile (g,q,kt) at constant offset q*65536 + g*4096 + kt*512.
    const __bf16* wb = Wp + (size_t)ng * 4096 + lane * 16;

    __syncthreads();

    for (int t = 0; t < T_; ++t) {
        const int cur = t & 1, nxt = cur ^ 1;

        v8f acc[2][4];
        #pragma unroll
        for (int g = 0; g < 2; ++g)
            #pragma unroll
            for (int q = 0; q < 4; ++q)
                acc[g][q] = (v8f){0.f,0.f,0.f,0.f,0.f,0.f,0.f,0.f};

        #pragma unroll 1
        for (int kt = 0; kt < 8; ++kt) {
            v16bf a = load_a(hb + mtile * 16 * H_, H_, kt * 32, lane);
            const __bf16* wk = wb + kt * 512;
            #pragma unroll
            for (int g = 0; g < 2; ++g) {
                v16bf bfr[4];
                #pragma unroll
                for (int q = 0; q < 4; ++q)
                    bfr[q] = *(const v16bf*)(wk + g * 4096 + q * 65536);
                #pragma unroll
                for (int q = 0; q < 4; ++q)
                    acc[g][q] = wmma_bf16(a, bfr[q], acc[g][q]);
            }
        }
        __syncthreads();   // all reads of hb done before in-place update

        // Stage x for t+1 (other buffer) while updating.  Re-read lengths[tid]
        // (L0-resident) so no thread-private scalar must survive the loop.
        if (tid < B_ && t + 1 < T_) {
            int tn = t + 1;
            int Lb = lengths[tid];
            int te = (dir == 0) ? tn : ((tn < Lb) ? (Lb - 1 - tn) : tn);
            const float* xp = x + ((size_t)tid * T_ + te) * 3;
            xs[nxt][tid][0] = xp[0];
            xs[nxt][tid][1] = xp[1];
            xs[nxt][tid][2] = xp[2];
        }

        #pragma unroll
        for (int g = 0; g < 2; ++g) {
            const int n = (ng + g) * 16 + ncol0;                 // 0..255
            v4f wq[4];
            #pragma unroll
            for (int q = 0; q < 4; ++q) wq[q] = wgb[q * H_ + n];
            #pragma unroll
            for (int r = 0; r < 8; ++r) {
                int brow = mtile * 16 + rowAdd + r;
                int Lr   = Lrow[r];
                bool on  = (t < Lr);
                int  te  = (dir == 0) ? t : (on ? (Lr - 1 - t) : t);
                float x0 = xs[cur][brow][0], x1 = xs[cur][brow][1], x2 = xs[cur][brow][2];
                float gi = acc[g][0][r] + wq[0].w + x0*wq[0].x + x1*wq[0].y + x2*wq[0].z;
                float gf = acc[g][1][r] + wq[1].w + x0*wq[1].x + x1*wq[1].y + x2*wq[1].z;
                float gg = acc[g][2][r] + wq[2].w + x0*wq[2].x + x1*wq[2].y + x2*wq[2].z;
                float go = acc[g][3][r] + wq[3].w + x0*wq[3].x + x1*wq[3].y + x2*wq[3].z;
                float cn = sigf(gf) * creg[g][r] + sigf(gi) * tanh_fast(gg);
                float hn = sigf(go) * tanh_fast(cn);
                float ho = on ? hn : 0.f;
                h1[((size_t)brow * T_ + te) * I1_ + dir * H_ + n] = (__bf16)ho;
                if (on) {
                    creg[g][r] = cn;
                    hb[brow * H_ + n] = (__bf16)hn;
                }
            }
        }
        __syncthreads();   // hb writes + xs[nxt] visible before next step
    }
}

// ---------------------------------------------------------------------------
// Grid-wide input-projection GEMM for layer-1 forward:
//   gx[t][mtile][nt] fragments = h1[:, t, :] @ W_ih(dir0)^T   (M=B*T, N=G, K=I1)
// Output stored directly in WMMA C/D fragment layout (f32) with NT hint
// (streamed once; keep weights/h1 resident in L2 instead).
// Block = (t, mtile); 8 waves, wave handles 8 N-tiles.
// ---------------------------------------------------------------------------
__global__ __launch_bounds__(256) void gx_gemm_kernel(
    const __bf16* __restrict__ h1,      // [B][T][512]
    const __bf16* __restrict__ Wp_ih,   // dir0 packed, KT=16
    float*        __restrict__ gx)      // [T*4][64][32][8] f32
{
    const int blk  = blockIdx.x;        // t*4 + mtile
    const int t    = blk >> 2;
    const int mtl  = blk & 3;
    const int wave = threadIdx.x >> 5;
    const int lane = threadIdx.x & 31;

    v8f acc[8];
    #pragma unroll
    for (int j = 0; j < 8; ++j)
        acc[j] = (v8f){0.f,0.f,0.f,0.f,0.f,0.f,0.f,0.f};

    const __bf16* asrc = h1 + (size_t)mtl * 16 * T_ * I1_ + (size_t)t * I1_;
    // wave's N-tile base: nt = wave*8 + j ; tile stride = 16 ktiles * 512 = 8192 elems
    const __bf16* wb = Wp_ih + (size_t)wave * 8 * 8192 + lane * 16;

    #pragma unroll 1
    for (int kt = 0; kt < 16; ++kt) {
        v16bf a = load_a(asrc, T_ * I1_, kt * 32, lane);
        const __bf16* wk = wb + kt * 512;
        #pragma unroll
        for (int h = 0; h < 2; ++h) {
            v16bf bfr[4];
            #pragma unroll
            for (int j = 0; j < 4; ++j)
                bfr[j] = *(const v16bf*)(wk + (h * 4 + j) * 8192);
            #pragma unroll
            for (int j = 0; j < 4; ++j)
                acc[h * 4 + j] = wmma_bf16(a, bfr[j], acc[h * 4 + j]);
        }
    }

    float* dst = gx + ((size_t)blk * 64 + wave * 8) * 256 + lane * 8;
    #pragma unroll
    for (int j = 0; j < 8; ++j)
        __builtin_nontemporal_store(acc[j], (v8f*)(dst + j * 256));
}

// ---------------------------------------------------------------------------
// Layer 1 forward recurrent LSTM (1 block, 1024 threads). If gx != nullptr the
// input projection was precomputed grid-wide (fast path: 8 recurrent k-tiles
// per step); otherwise it is computed inline (16 extra k-tiles per step).
// Only the final masked h is needed -> feat[:, 0:256].
// ---------------------------------------------------------------------------
__global__ __launch_bounds__(1024) void lstm_l1_fwd_kernel(
    const __bf16* __restrict__ h1,      // [B][T][512]
    const int*    __restrict__ lengths,
    const __bf16* __restrict__ Wp_ih,   // dir0 packed, KT=16
    const __bf16* __restrict__ Wp_hh,   // dir0 packed, KT=8
    const float*  __restrict__ bias,    // dir0 [G]
    const float*  __restrict__ gx,      // fragment-layout gx or nullptr
    float*        __restrict__ feat)    // [B][512]
{
    const int tid  = threadIdx.x;
    const int wave = tid >> 5;
    const int lane = tid & 31;
    const int mtl  = wave >> 3;
    const int ng   = (wave & 7) * 2;

    __shared__ __align__(32) __bf16 hb[B_ * H_];
    __shared__ float bbs[G_];

    for (int i = tid; i < B_ * H_; i += 1024) hb[i] = (__bf16)0.0f;
    for (int j = tid; j < G_; j += 1024) bbs[j] = bias[j];

    float creg[2][8];
    #pragma unroll
    for (int g = 0; g < 2; ++g)
        #pragma unroll
        for (int r = 0; r < 8; ++r) creg[g][r] = 0.0f;

    const int ncol0  = lane & 15;
    const int rowAdd = (lane >> 4) << 3;

    int Lrow[8];
    #pragma unroll
    for (int r = 0; r < 8; ++r) Lrow[r] = lengths[mtl * 16 + rowAdd + r];

    const __bf16* wbih = Wp_ih + (size_t)ng * 8192 + lane * 16;  // kt 512, g 8192, q 131072
    const __bf16* wbhh = Wp_hh + (size_t)ng * 4096 + lane * 16;  // kt 512, g 4096, q 65536

    __syncthreads();                      // hb/bbs init visible

    for (int t = 0; t < T_; ++t) {
        v8f acc[2][4];
        if (gx) {
            // Accumulator init = precomputed input projection fragments (NT).
            const float* gbase = gx + (size_t)(t * 4 + mtl) * 64 * 256 + lane * 8;
            #pragma unroll
            for (int g = 0; g < 2; ++g)
                #pragma unroll
                for (int q = 0; q < 4; ++q)
                    acc[g][q] = __builtin_nontemporal_load(
                        (const v8f*)(gbase + (size_t)(q * 16 + ng + g) * 256));
            if (t + 1 < T_)
                __builtin_prefetch(gbase + (size_t)4 * 64 * 256, 0, 3);
        } else {
            #pragma unroll
            for (int g = 0; g < 2; ++g)
                #pragma unroll
                for (int q = 0; q < 4; ++q)
                    acc[g][q] = (v8f){0.f,0.f,0.f,0.f,0.f,0.f,0.f,0.f};
            const __bf16* asrc = h1 + (size_t)mtl * 16 * T_ * I1_ + (size_t)t * I1_;
            #pragma unroll 1
            for (int kt = 0; kt < 16; ++kt) {
                v16bf a = load_a(asrc, T_ * I1_, kt * 32, lane);
                const __bf16* wk = wbih + kt * 512;
                #pragma unroll
                for (int g = 0; g < 2; ++g) {
                    v16bf bfr[4];
                    #pragma unroll
                    for (int q = 0; q < 4; ++q)
                        bfr[q] = *(const v16bf*)(wk + g * 8192 + q * 131072);
                    #pragma unroll
                    for (int q = 0; q < 4; ++q)
                        acc[g][q] = wmma_bf16(a, bfr[q], acc[g][q]);
                }
            }
        }

        // Recurrent part from LDS h.
        #pragma unroll 1
        for (int kt = 0; kt < 8; ++kt) {
            v16bf a = load_a(hb + mtl * 16 * H_, H_, kt * 32, lane);
            const __bf16* wk = wbhh + kt * 512;
            #pragma unroll
            for (int g = 0; g < 2; ++g) {
                v16bf bfr[4];
                #pragma unroll
                for (int q = 0; q < 4; ++q)
                    bfr[q] = *(const v16bf*)(wk + g * 4096 + q * 65536);
                #pragma unroll
                for (int q = 0; q < 4; ++q)
                    acc[g][q] = wmma_bf16(a, bfr[q], acc[g][q]);
            }
        }
        __syncthreads();   // hb reads complete before in-place update

        #pragma unroll
        for (int g = 0; g < 2; ++g) {
            const int n = (ng + g) * 16 + ncol0;
            float b0 = bbs[n], b1 = bbs[H_ + n], b2 = bbs[2*H_ + n], b3 = bbs[3*H_ + n];
            #pragma unroll
            for (int r = 0; r < 8; ++r) {
                int brow = mtl * 16 + rowAdd + r;
                if (t < Lrow[r]) {
                    float cn = sigf(acc[g][1][r] + b1) * creg[g][r]
                             + sigf(acc[g][0][r] + b0) * tanh_fast(acc[g][2][r] + b2);
                    float hn = sigf(acc[g][3][r] + b3) * tanh_fast(cn);
                    creg[g][r] = cn;
                    hb[brow * H_ + n] = (__bf16)hn;
                }
            }
        }
        __syncthreads();   // hb writes visible before next step's reads
    }
    for (int i = tid; i < B_ * H_; i += 1024) {
        int b = i / H_, n = i % H_;
        feat[b * I1_ + n] = (float)hb[i];
    }
}

// ---------------------------------------------------------------------------
// Layer 1 backward direction: exactly ONE step from zero state on input
// h1[b, L-1] (since feat_bwd = bwd_r[:, 0]). Scalar, grid-parallel (67 MFLOP).
// ---------------------------------------------------------------------------
__global__ __launch_bounds__(256) void lstm_l1_bwd1_kernel(
    const __bf16* __restrict__ h1,
    const int*    __restrict__ lengths,
    const float*  __restrict__ W_ih1,   // dir1: [G][512] (f32, unpacked)
    const float*  __restrict__ bias,    // dir1 [G]
    float*        __restrict__ feat)
{
    int b = blockIdx.x;
    int n = threadIdx.x;
    int L = lengths[b];
    const __bf16* hr = h1 + ((size_t)b * T_ + (L - 1)) * I1_;
    float g4[4];
    #pragma unroll
    for (int q = 0; q < 4; ++q) {
        const float* w = W_ih1 + (size_t)(q * H_ + n) * I1_;
        float s = bias[q * H_ + n];
        for (int k = 0; k < I1_; ++k) s += (float)hr[k] * w[k];
        g4[q] = s;
    }
    float cn = sigf(g4[0]) * tanh_fast(g4[2]);   // f-gate * 0 + i * tanh(g)
    float hn = sigf(g4[3]) * tanh_fast(cn);
    feat[b * I1_ + H_ + n] = hn;
}

// ---------------------------------------------------------------------------
// Head: y = relu(feat @ fc1_w^T + b); batchnorm over batch (biased var);
// out = y @ fc_out_w^T + b.  One block, 256 threads (thread j owns column j).
// ---------------------------------------------------------------------------
__global__ __launch_bounds__(256) void head_kernel(
    const float* __restrict__ feat,   // [64][512]
    const float* __restrict__ fc1_w,  // [256][512]
    const float* __restrict__ fc1_b,
    const float* __restrict__ gamma,
    const float* __restrict__ beta,
    const float* __restrict__ fo_w,   // [196][256]
    const float* __restrict__ fo_b,
    float* __restrict__ out)          // [64][196]
{
    __shared__ float y[B_][256];
    int j = threadIdx.x;
    float col[B_];
    const float* w = fc1_w + (size_t)j * I1_;
    for (int b = 0; b < B_; ++b) {
        float s = fc1_b[j];
        for (int k = 0; k < I1_; ++k) s += feat[b * I1_ + k] * w[k];
        col[b] = s > 0.f ? s : 0.f;
    }
    float mean = 0.f;
    for (int b = 0; b < B_; ++b) mean += col[b];
    mean *= (1.0f / B_);
    float var = 0.f;
    for (int b = 0; b < B_; ++b) { float d = col[b] - mean; var += d * d; }
    var *= (1.0f / B_);
    float scale = gamma[j] * __frsqrt_rn(var + 1e-5f);
    float bb = beta[j];
    for (int b = 0; b < B_; ++b) y[b][j] = (col[b] - mean) * scale + bb;
    __syncthreads();
    for (int idx = j; idx < B_ * 196; idx += 256) {
        int b = idx / 196, o = idx % 196;
        const float* wo = fo_w + (size_t)o * 256;
        float s = fo_b[o];
        for (int k = 0; k < 256; ++k) s += y[b][k] * wo[k];
        out[b * 196 + o] = s;
    }
}

// ---------------------------------------------------------------------------
extern "C" void kernel_launch(void* const* d_in, const int* in_sizes, int n_in,
                              void* d_out, int out_size, void* d_ws, size_t ws_size,
                              hipStream_t stream) {
    const float* x        = (const float*)d_in[0];
    const int*   lengths  = (const int*)  d_in[1];
    const float* W_ih_l0  = (const float*)d_in[2];
    const float* W_hh_l0  = (const float*)d_in[3];
    const float* b_ih_l0  = (const float*)d_in[4];
    const float* b_hh_l0  = (const float*)d_in[5];
    const float* W_ih_l1  = (const float*)d_in[6];
    const float* W_hh_l1  = (const float*)d_in[7];
    const float* b_ih_l1  = (const float*)d_in[8];
    const float* b_hh_l1  = (const float*)d_in[9];
    const float* fc1_w    = (const float*)d_in[10];
    const float* fc1_b    = (const float*)d_in[11];
    const float* bn_g     = (const float*)d_in[12];
    const float* bn_b     = (const float*)d_in[13];
    const float* fo_w     = (const float*)d_in[14];
    const float* fo_b     = (const float*)d_in[15];
    float* out = (float*)d_out;

    char* ws = (char*)d_ws;
    size_t off = 0;
    auto alloc = [&](size_t bytes) -> void* {
        void* p = ws + off;
        off = (off + bytes + 255) & ~(size_t)255;
        return p;
    };
    __bf16* Wp_hh0 = (__bf16*)alloc((size_t)2 * G_ * H_  * 2);
    __bf16* Wp_hh1 = (__bf16*)alloc((size_t)2 * G_ * H_  * 2);
    __bf16* Wp_ih1 = (__bf16*)alloc((size_t)2 * G_ * I1_ * 2);
    float*  bias0  = (float*) alloc((size_t)2 * G_ * 4);
    float*  bias1  = (float*) alloc((size_t)2 * G_ * 4);
    float*  feat   = (float*) alloc((size_t)B_ * I1_ * 4);
    __bf16* h1     = (__bf16*)alloc((size_t)B_ * T_ * I1_ * 2);   // 64 MB

    // Optional 256 MB buffer for the grid-parallel layer-1 input projection.
    const size_t gx_bytes = (size_t)T_ * 4 * 64 * 256 * 4;
    float* gx = nullptr;
    if (ws_size >= off + gx_bytes) gx = (float*)alloc(gx_bytes);

    int tot;
    tot = 2 * G_ * H_;
    pack_b_kernel<<<(tot + 255) / 256, 256, 0, stream>>>(W_hh_l0, Wp_hh0, H_, tot);
    pack_b_kernel<<<(tot + 255) / 256, 256, 0, stream>>>(W_hh_l1, Wp_hh1, H_, tot);
    tot = 2 * G_ * I1_;
    pack_b_kernel<<<(tot + 255) / 256, 256, 0, stream>>>(W_ih_l1, Wp_ih1, I1_, tot);
    bias_kernel<<<16, 256, 0, stream>>>(b_ih_l0, b_hh_l0, b_ih_l1, b_hh_l1, bias0, bias1);

    lstm_l0_kernel<<<2, 1024, 0, stream>>>(x, lengths, W_ih_l0, Wp_hh0, bias0, h1);
    if (gx)
        gx_gemm_kernel<<<T_ * 4, 256, 0, stream>>>(h1, Wp_ih1, gx);
    lstm_l1_fwd_kernel<<<1, 1024, 0, stream>>>(h1, lengths, Wp_ih1, Wp_hh1, bias1, gx, feat);
    lstm_l1_bwd1_kernel<<<B_, 256, 0, stream>>>(h1, lengths,
                                                W_ih_l1 + (size_t)G_ * I1_,
                                                bias1 + G_, feat);
    head_kernel<<<1, 256, 0, stream>>>(feat, fc1_w, fc1_b, bn_g, bn_b, fo_w, fo_b, out);
}
